// DkNNModel_13743895347613
// MI455X (gfx1250) — compile-verified, matrix operand order checked by reading, not run on hardware
//
#include <hip/hip_runtime.h>
#include <hip/hip_bf16.h>

// ---------------------------------------------------------------------------
// DkNN on MI455X (gfx1250, wave32).
//
// Math reduction: centering cancels ->  d2(q,x) = 2 - 2 * (qn . xn).
// Cosine GEMM S = Qn * Xn^T (1024 x 100000, D=256) fused with per-query
// top-75 selection; the 400MB score matrix is never materialized.
//
// Precision: fp32 -> (f16 hi + f16 lo) split, s ~= qh.xh + qh.xl + ql.xh
// (drop ql.xl, ~2^-22 error) on V_WMMA_F32_16X16X32_F16. Norm reciprocals
// are folded into the data BEFORE the split, once, in a prep kernel.
//
// Data movement: pre-split f16 hi/lo copies of Q and X live in workspace;
// GEMM tiles are staged into LDS by the Tensor Data Mover
// (tensor_load_to_lds + s_wait_tensorcnt), with D# padding producing the
// bank-friendly 144B LDS row stride directly.
// ---------------------------------------------------------------------------

#define K_NN 75
#define NB_CLASSES 10
#define NQ 1024
#define DIM 256
#define NTRAIN 100000
#define NSEG 8
#define SEGLEN (NTRAIN / NSEG)              // 12500
#define SEG_CHUNKS ((SEGLEN + 15) / 16)     // 782 (last chunk: 4 valid rows)
#define QTILE 16
#define NQT (NQ / QTILE)                    // 64

#define QSTR 264   // f16 LDS row stride for Q tiles (512B + 16B pad)
#define XSTR 72    // f16 LDS row stride for X tiles (128B + 16B pad)

typedef __attribute__((ext_vector_type(16))) _Float16 v16h;
typedef __attribute__((ext_vector_type(8)))  _Float16 v8h;
typedef __attribute__((ext_vector_type(8)))  float    v8f;
typedef __attribute__((ext_vector_type(4)))  unsigned int u32x4;
typedef __attribute__((ext_vector_type(8)))  int i32x8;
typedef __attribute__((ext_vector_type(4)))  int i32x4;

#define FLT_BIG 3.0e38f

#if defined(__has_builtin)
#if __has_builtin(__builtin_amdgcn_tensor_load_to_lds)
#define DKNN_HAVE_TDM 1
#endif
#endif

#ifdef DKNN_HAVE_TDM
// Issue one 2D tensor_load_to_lds. Units are data_size = 8 bytes.
//  - pad_int / pad_amt are the encoded D# fields: LDS gets pad_amt padding
//    after every (1<<pad_int) 8B-units stored (gives padded LDS row stride).
//  - tens_d1 < tile_d1 rows => hardware returns zeros (OOB clamp at array end).
__device__ __forceinline__ void tdm_load_2d(const void* gptr, unsigned ldsoff,
                                            unsigned tens_d0, unsigned tens_d1,
                                            unsigned tile_d0, unsigned tile_d1,
                                            unsigned stride0,
                                            unsigned pad_int, unsigned pad_amt) {
  unsigned long long ga = (unsigned long long)(uintptr_t)gptr;
  u32x4 g0;
  g0[0] = 1u;                                   // count=1, user mode, no gather
  g0[1] = ldsoff;                               // LDS byte address
  g0[2] = (unsigned)ga;                         // global_addr[31:0]
  g0[3] = (unsigned)((ga >> 32) & 0x01FFFFFFu)  // global_addr[56:32]
        | 0x80000000u;                          // type=2 ("image")
  i32x8 g1;
  g1[0] = (int)((3u << 16)                      // data_size = 8B
              | (1u << 20)                      // pad_enable
              | ((pad_int & 7u) << 22)          // pad_interval code
              | ((pad_amt & 127u) << 25));      // pad_amount code
  g1[1] = (int)((tens_d0 & 0xFFFFu) << 16);                              // tensor_dim0[15:0]
  g1[2] = (int)(((tens_d0 >> 16) & 0xFFFFu) | ((tens_d1 & 0xFFFFu) << 16));
  g1[3] = (int)(((tens_d1 >> 16) & 0xFFFFu) | ((tile_d0 & 0xFFFFu) << 16));
  g1[4] = (int)(tile_d1 & 0xFFFFu);             // tile_dim1, tile_dim2=0 (2D)
  g1[5] = (int)stride0;                         // tensor_dim0_stride[31:0]
  g1[6] = 0;                                    // stride0[47:32], stride1 lo
  g1[7] = 0;                                    // stride1 hi (unused, 2D)
  i32x4 z4 = {0, 0, 0, 0};
#if defined(__clang_major__) && __clang_major__ >= 23
  i32x8 z8 = {0, 0, 0, 0, 0, 0, 0, 0};
  __builtin_amdgcn_tensor_load_to_lds(g0, g1, z4, z4, z8, 0);
#else
  __builtin_amdgcn_tensor_load_to_lds(g0, g1, z4, z4, 0);
#endif
}
#endif  // DKNN_HAVE_TDM

// Load a 16-half WMMA A/B fragment from LDS: two aligned 16B runs
// (K = base..base+7 and base+16..base+23 for this lane's K-half).
__device__ __forceinline__ v16h load_frag_lds(const _Float16* p) {
  const v8h* p0 = (const v8h*)p;
  const v8h* p1 = (const v8h*)(p + 16);
  v16h r;
#pragma unroll
  for (int i = 0; i < 8; ++i) { r[i] = (*p0)[i]; r[8 + i] = (*p1)[i]; }
  return r;
}

// ---------------------------------------------------------------------------
// Kernel 1: fused norms + f16 hi/lo pre-split (one wave per row).
// Reads each row once: sum of squares -> rsqrt -> scaled split -> f16 arrays.
// ---------------------------------------------------------------------------
__global__ void dknn_prep_kernel(const float* __restrict__ queries,
                                 const float* __restrict__ train,
                                 _Float16* __restrict__ qhi,
                                 _Float16* __restrict__ qlo,
                                 _Float16* __restrict__ xhi,
                                 _Float16* __restrict__ xlo) {
  const int w = threadIdx.x >> 5;
  const int lane = threadIdx.x & 31;
  const long long row = (long long)blockIdx.x * 8 + w;
  if (row >= (long long)(NQ + NTRAIN)) return;
  const float* src;
  _Float16 *dhi, *dlo;
  if (row < NQ) {
    src = queries + row * DIM; dhi = qhi + row * DIM; dlo = qlo + row * DIM;
  } else {
    const long long r = row - NQ;
    src = train + r * DIM; dhi = xhi + r * DIM; dlo = xlo + r * DIM;
  }
  float v[8];
  float s = 0.f;
#pragma unroll
  for (int j = 0; j < 4; ++j) {           // pairs: elems 2*lane + 64*j (+1)
    v[2 * j + 0] = src[2 * lane + 64 * j + 0];
    v[2 * j + 1] = src[2 * lane + 64 * j + 1];
    s = fmaf(v[2 * j], v[2 * j], s);
    s = fmaf(v[2 * j + 1], v[2 * j + 1], s);
  }
#pragma unroll
  for (int off = 16; off > 0; off >>= 1) s += __shfl_xor(s, off, 32);
  const float r = rsqrtf(s);
#pragma unroll
  for (int j = 0; j < 4; ++j) {
    union { _Float16 h[2]; unsigned u; } ph, pl;
    float f0 = v[2 * j] * r, f1 = v[2 * j + 1] * r;
    ph.h[0] = (_Float16)f0; ph.h[1] = (_Float16)f1;
    pl.h[0] = (_Float16)(f0 - (float)ph.h[0]);
    pl.h[1] = (_Float16)(f1 - (float)ph.h[1]);
    ((unsigned*)dhi)[lane + 32 * j] = ph.u;
    ((unsigned*)dlo)[lane + 32 * j] = pl.u;
  }
}

// ---------------------------------------------------------------------------
// Kernel 2: fused WMMA cosine-GEMM + streaming per-query top-K.
// Grid: (64 query-tiles of 16, 8 train segments). Block: 256 = 8 waves.
// Each wave owns every 8th 16-row train chunk of its segment.
// ---------------------------------------------------------------------------
__global__ void __launch_bounds__(256)
dknn_gemm_topk_kernel(const _Float16* __restrict__ qhi_g,
                      const _Float16* __restrict__ qlo_g,
                      const _Float16* __restrict__ xhi_g,
                      const _Float16* __restrict__ xlo_g,
                      float* __restrict__ mdist,
                      int* __restrict__ midx) {
  __shared__ __align__(16) _Float16 qhiH[QTILE * QSTR];   //  8448 B
  __shared__ __align__(16) _Float16 qloH[QTILE * QSTR];   //  8448 B
  __shared__ __align__(16) _Float16 xhiH[8][16 * XSTR];   // 18432 B
  __shared__ __align__(16) _Float16 xloH[8][16 * XSTR];   // 18432 B
  __shared__ float stageS[8][16][16];                     //  8192 B
  __shared__ float ldistS[8][QTILE][K_NN];                // 38400 B
  __shared__ int   lidxS [8][QTILE][K_NN];                // 38400 B
  // total 138752 B (< 320 KB WGP LDS)

  const int w = threadIdx.x >> 5;
  const int lane = threadIdx.x & 31;
  const int q0 = blockIdx.x * QTILE;
  const int n_start = blockIdx.y * SEGLEN;

  // ---- stage Q tile (16 x 256 halves, LDS stride QSTR) ----
#ifdef DKNN_HAVE_TDM
  if (w == 0) {
    // 16 rows x 512B, pad 16B after every 512B -> LDS stride 528B (= QSTR)
    tdm_load_2d(qhi_g + (size_t)q0 * DIM, (unsigned)(uintptr_t)&qhiH[0],
                /*tens_d0=*/64, /*tens_d1=*/16, /*tile_d0=*/64, /*tile_d1=*/16,
                /*stride0=*/64, /*pad_int=*/6, /*pad_amt=*/3);
    tdm_load_2d(qlo_g + (size_t)q0 * DIM, (unsigned)(uintptr_t)&qloH[0],
                64, 16, 64, 16, 64, 6, 3);
    asm volatile("" ::: "memory");
    __builtin_amdgcn_s_wait_tensorcnt(0);
    asm volatile("" ::: "memory");
  }
#else
  {
    const int row = threadIdx.x >> 4;        // 0..15
    const int cb  = (threadIdx.x & 15) * 16; // halves 0..240
    const uint4* sh = (const uint4*)(qhi_g + (size_t)(q0 + row) * DIM + cb);
    const uint4* sl = (const uint4*)(qlo_g + (size_t)(q0 + row) * DIM + cb);
    uint4* dh = (uint4*)(qhiH + row * QSTR + cb);
    uint4* dl = (uint4*)(qloH + row * QSTR + cb);
    dh[0] = sh[0]; dh[1] = sh[1];
    dl[0] = sl[0]; dl[1] = sl[1];
  }
#endif
  // ---- init this wave's top-K lists ----
  for (int j = lane; j < QTILE * K_NN; j += 32) {
    (&ldistS[w][0][0])[j] = FLT_BIG;
    (&lidxS [w][0][0])[j] = 0;
  }
  __syncthreads();

  const int mrow  = lane & 15;        // A row / B col owned by this lane
  const int khalf = (lane >> 4) & 1;  // which K-half this lane holds
  const int mbase = khalf * 8;        // C rows (queries) held by this lane

  // per-lane streaming top-K state (lane q < 16 owns query q's list)
  int   cnt = 0;
  float curmax = -FLT_BIG;
  int   maxpos = 0;

  for (int chunk = w; chunk < SEG_CHUNKS; chunk += 8) {
    const int nb_local = chunk * 16;
    const int nb = n_start + nb_local;
    const int rows_valid = min(16, SEGLEN - nb_local);
    v8f acc = {};

    for (int kg = 0; kg < DIM; kg += 64) {
#ifdef DKNN_HAVE_TDM
      {
        // LDS buffer may still feed in-flight ds_loads of the previous
        // k-group; drain DS before the TDM engine overwrites it.
        asm volatile("s_wait_dscnt 0x0" ::: "memory");
        const unsigned rows_avail = (unsigned)min(16, NTRAIN - nb);
        // 16 rows x 128B, pad 16B after every 128B -> LDS stride 144B (=XSTR)
        tdm_load_2d(xhi_g + (size_t)nb * DIM + kg,
                    (unsigned)(uintptr_t)&xhiH[w][0],
                    /*tens_d0=*/16, /*tens_d1=*/rows_avail,
                    /*tile_d0=*/16, /*tile_d1=*/16,
                    /*stride0=*/64, /*pad_int=*/4, /*pad_amt=*/3);
        tdm_load_2d(xlo_g + (size_t)nb * DIM + kg,
                    (unsigned)(uintptr_t)&xloH[w][0],
                    16, rows_avail, 16, 16, 64, 4, 3);
        asm volatile("" ::: "memory");
        __builtin_amdgcn_s_wait_tensorcnt(0);
        asm volatile("" ::: "memory");
      }
#else
      {
        const int row = lane >> 1;
        const int grow = min(nb + row, NTRAIN - 1);
        const size_t gb = (size_t)grow * DIM + kg + (lane & 1) * 32;
        const uint4* sh = (const uint4*)(xhi_g + gb);
        const uint4* sl = (const uint4*)(xlo_g + gb);
        uint4* dh = (uint4*)(&xhiH[w][0] + row * XSTR + (lane & 1) * 32);
        uint4* dl = (uint4*)(&xloH[w][0] + row * XSTR + (lane & 1) * 32);
#pragma unroll
        for (int j = 0; j < 4; ++j) { dh[j] = sh[j]; dl[j] = sl[j]; }
      }
      __builtin_amdgcn_wave_barrier();  // per-wave LDS ops are in-order
#endif

#pragma unroll
      for (int ks2 = 0; ks2 < 2; ++ks2) {
        const int qa = mrow * QSTR + kg + ks2 * 32 + khalf * 8;
        const int xb = mrow * XSTR + ks2 * 32 + khalf * 8;
        v16h ah = load_frag_lds(qhiH + qa);
        v16h al = load_frag_lds(qloH + qa);
        v16h bh = load_frag_lds(&xhiH[w][0] + xb);
        v16h bl = load_frag_lds(&xloH[w][0] + xb);
        acc = __builtin_amdgcn_wmma_f32_16x16x32_f16(false, ah, false, bh,
                                                     (short)0, acc, false, false);
        acc = __builtin_amdgcn_wmma_f32_16x16x32_f16(false, ah, false, bl,
                                                     (short)0, acc, false, false);
        acc = __builtin_amdgcn_wmma_f32_16x16x32_f16(false, al, false, bh,
                                                     (short)0, acc, false, false);
      }
      __builtin_amdgcn_wave_barrier();
    }

    // d2 = 2 - 2*cos; transpose through LDS for per-query insertion
#pragma unroll
    for (int i = 0; i < 8; ++i)
      stageS[w][mbase + i][mrow] = fmaf(-2.f, acc[i], 2.f);
    __builtin_amdgcn_wave_barrier();

    if (lane < 16) {
#pragma unroll 1
      for (int c = 0; c < 16; ++c) {
        if (c >= rows_valid) break;
        const float d = stageS[w][lane][c];
        const int id = nb + c;
        if (cnt < K_NN) {
          ldistS[w][lane][cnt] = d; lidxS[w][lane][cnt] = id;
          if (d > curmax) { curmax = d; maxpos = cnt; }
          ++cnt;
        } else if (d < curmax) {                 // rare: replace current max
          ldistS[w][lane][maxpos] = d; lidxS[w][lane][maxpos] = id;
          float m = -FLT_BIG; int mp = 0;
#pragma unroll 1
          for (int jj = 0; jj < K_NN; ++jj) {
            const float vv = ldistS[w][lane][jj];
            if (vv > m) { m = vv; mp = jj; }
          }
          curmax = m; maxpos = mp;
        }
      }
    }
    __builtin_amdgcn_wave_barrier();

    if (chunk + 8 < SEG_CHUNKS) {               // global_prefetch_b8 (L2 warm)
      __builtin_prefetch(xhi_g + (size_t)(nb + 128) * DIM, 0, 1);
      __builtin_prefetch(xlo_g + (size_t)(nb + 128) * DIM, 0, 1);
    }
  }

#ifdef DKNN_HAVE_TDM
  asm volatile("s_wait_dscnt 0x0" ::: "memory");
#endif
  __syncthreads();

  // ---- merge 8 per-wave lists -> one ascending 75-list per (query,segment)
  // wave w merges queries 2w and 2w+1 via wave-parallel argmin selection.
  for (int qq = 0; qq < 2; ++qq) {
    const int q = 2 * w + qq;
    for (int t = 0; t < K_NN; ++t) {
      float best = FLT_BIG; int bj = 0x7fffffff;
      for (int j = lane; j < 8 * K_NN; j += 32) {
        const float v = ldistS[j / K_NN][q][j % K_NN];
        if (v < best) { best = v; bj = j; }
      }
#pragma unroll
      for (int off = 16; off > 0; off >>= 1) {
        const float ov = __shfl_xor(best, off, 32);
        const int   oj = __shfl_xor(bj, off, 32);
        if (ov < best || (ov == best && oj < bj)) { best = ov; bj = oj; }
      }
      if (lane == 0) {
        const int sw = bj / K_NN, e = bj % K_NN;
        ldistS[sw][q][e] = FLT_BIG;              // consume
        const size_t o = ((size_t)blockIdx.y * NQ + (q0 + q)) * K_NN + t;
        mdist[o] = best;
        midx[o]  = lidxS[sw][q][e];
      }
      __builtin_amdgcn_wave_barrier();
    }
  }
}

// ---------------------------------------------------------------------------
// Kernel 3: per query, 8-way merge of sorted segment lists -> final top-75,
// sqrt distances, neighbor indices, 10-class histogram. One wave per query.
// ---------------------------------------------------------------------------
__global__ void dknn_final_kernel(const float* __restrict__ mdist,
                                  const int* __restrict__ midx,
                                  const int* __restrict__ labels,
                                  float* __restrict__ out_d,
                                  int* __restrict__ out_i,
                                  float* __restrict__ out_c) {
  __shared__ float fd[K_NN];
  __shared__ int   fi[K_NN];
  __shared__ int   hist[NB_CLASSES];
  const int q = blockIdx.x;
  const int lane = threadIdx.x;
  if (lane < NB_CLASSES) hist[lane] = 0;
  if (lane == 0) {
    int h[NSEG];
#pragma unroll
    for (int s = 0; s < NSEG; ++s) h[s] = 0;
    for (int t = 0; t < K_NN; ++t) {
      float best = FLT_BIG; int bs = 0;
#pragma unroll
      for (int s = 0; s < NSEG; ++s) {
        if (h[s] < K_NN) {
          const float v = mdist[((size_t)s * NQ + q) * K_NN + h[s]];
          if (v < best) { best = v; bs = s; }
        }
      }
      fd[t] = best;
      fi[t] = midx[((size_t)bs * NQ + q) * K_NN + h[bs]];
      h[bs]++;
    }
  }
  __syncthreads();
  for (int t = lane; t < K_NN; t += 32) {
    const float d2 = fd[t];
    out_d[(size_t)q * K_NN + t] = sqrtf(fmaxf(d2, 0.f));
    const int id = fi[t];
    out_i[(size_t)q * K_NN + t] = id;
    atomicAdd(&hist[labels[id]], 1);
  }
  __syncthreads();
  if (lane < NB_CLASSES)
    out_c[(size_t)q * NB_CLASSES + lane] = (float)hist[lane];
}

// ---------------------------------------------------------------------------
extern "C" void kernel_launch(void* const* d_in, const int* in_sizes, int n_in,
                              void* d_out, int out_size, void* d_ws, size_t ws_size,
                              hipStream_t stream) {
  (void)in_sizes; (void)n_in; (void)out_size; (void)ws_size;
  const float* queries = (const float*)d_in[0];   // [1024, 256] f32
  const float* train   = (const float*)d_in[1];   // [100000, 256] f32
  const int*   labels  = (const int*)d_in[2];     // [100000] i32

  // workspace layout (~112 MB assumed available):
  //   qhi @ 0x0        (512 KB)   qlo @ 0x80000    (512 KB)
  //   mdist @ 0x100000 (2.34 MB)  midx @ 0x380000  (2.34 MB)
  //   xhi @ 0x600000   (48.8 MB)  xlo @ 0x3800000  (48.8 MB)
  char* ws = (char*)d_ws;
  _Float16* qhi  = (_Float16*)(ws);
  _Float16* qlo  = (_Float16*)(ws + 0x80000);
  float*    mdist = (float*)(ws + 0x100000);
  int*      midx  = (int*)  (ws + 0x380000);
  _Float16* xhi  = (_Float16*)(ws + 0x600000);
  _Float16* xlo  = (_Float16*)(ws + 0x3800000);

  float* out_d = (float*)d_out;                     // [1024*75] distances
  int*   out_i = (int*)((float*)d_out + NQ * K_NN); // [1024*75] neighbor idx
  float* out_c = (float*)d_out + 2 * NQ * K_NN;     // [1024*10] class counts

  dknn_prep_kernel<<<dim3((NQ + NTRAIN + 7) / 8), 256, 0, stream>>>(
      queries, train, qhi, qlo, xhi, xlo);
  dknn_gemm_topk_kernel<<<dim3(NQT, NSEG), 256, 0, stream>>>(
      qhi, qlo, xhi, xlo, mdist, midx);
  dknn_final_kernel<<<dim3(NQ), 32, 0, stream>>>(
      mdist, midx, labels, out_d, out_i, out_c);
}
